// TLSTM_PyTorch_49254684950841
// MI455X (gfx1250) — matrix operand order repaired and checked by reading, not compiled, not run on hardware
//
#include <hip/hip_runtime.h>
#include <hip/hip_bf16.h>

// ---------------------------------------------------------------------------
// T-LSTM for MI455X (gfx1250): f16 WMMA (V_WMMA_F32_16X16X32_F16) GEMMs with
// f32 accumulation. Per time step, ONE merged kernel runs both independent
// recurrent GEMMs concurrently (gate GEMM fused as [h | x_t] @ [Ux ; Wx]),
// followed by one pointwise kernel -> 2 dependent launches per step.
// ---------------------------------------------------------------------------

typedef __attribute__((ext_vector_type(16))) _Float16 v16h;
typedef __attribute__((ext_vector_type(8)))  _Float16 v8h;
typedef __attribute__((ext_vector_type(8)))  float    v8f;

#define B_   256
#define T_   256
#define DIN  256
#define H_   1024
#define G4   4096      // 4*H
#define KC   1280      // H + DIN (fused K for gate GEMM)
#define FC_  512
#define DOUT 24

#define GATE_BLOCKS 128   // (256/16 mTiles) * (4096/64 n-strips) / 8 waves
#define DEC_BLOCKS  32    // (256/16 mTiles) * (1024/64 n-strips) / 8 waves

static __device__ __forceinline__ v8f wmma32(v16h a, v16h b, v8f c) {
  // D = A(16x32 f16) x B(32x16 f16) + C(16x16 f32)
  return __builtin_amdgcn_wmma_f32_16x16x32_f16(
      /*neg_a=*/false, a, /*neg_b=*/false, b,
      /*c_mod=*/(short)0, c, /*reuse_a=*/false, /*reuse_b=*/false);
}

static __device__ __forceinline__ float sigm(float x) {
  return 1.0f / (1.0f + expf(-x));
}

// Build a v16h A-fragment for lane group `grp`, contiguous K order (matches B):
// elements 0..7  -> K = k + grp*16 + {0..7}
// elements 8..15 -> K = k + grp*16 + 8 + {0..7}
static __device__ __forceinline__ v16h load_a_frag(const _Float16* __restrict__ ap, int grp) {
  v8h lo = *(const v8h*)(ap + grp * 16);
  v8h hi = *(const v8h*)(ap + grp * 16 + 8);
  return __builtin_shufflevector(lo, hi, 0,1,2,3,4,5,6,7,8,9,10,11,12,13,14,15);
}

// ---------------------------------------------------------------------------
// Precompute / conversion kernels (one-time per launch)
// ---------------------------------------------------------------------------

__global__ void k_f32_to_f16(const float* __restrict__ src, _Float16* __restrict__ dst, int n) {
  int i = blockIdx.x * blockDim.x + threadIdx.x;
  if (i < n) dst[i] = (_Float16)src[i];
}

// BcombT[n][k] (f16, [4096][1280]) = k<1024 ? Ux[k][n] : Wx[k-1024][n]
__global__ void k_build_bcomb(const float* __restrict__ Ux, const float* __restrict__ Wx,
                              _Float16* __restrict__ dst) {
  int idx = blockIdx.x * blockDim.x + threadIdx.x;
  if (idx >= G4 * KC) return;
  int n = idx / KC;
  int k = idx % KC;
  float v = (k < H_) ? Ux[(size_t)k * G4 + n] : Wx[(size_t)(k - H_) * G4 + n];
  dst[idx] = (_Float16)v;
}

// dst[n][k] (f16, [N][K]) = src[k][n] (f32, [K][N])
__global__ void k_transpose_f16(const float* __restrict__ src, _Float16* __restrict__ dst,
                                int K, int N) {
  int idx = blockIdx.x * blockDim.x + threadIdx.x;
  if (idx >= K * N) return;
  int k = idx / N;
  int n = idx % N;
  dst[(size_t)n * K + k] = (_Float16)src[idx];
}

__global__ void k_zero_state(float* __restrict__ c, float* __restrict__ h,
                             _Float16* __restrict__ c16, _Float16* __restrict__ h16, int n) {
  int i = blockIdx.x * blockDim.x + threadIdx.x;
  if (i < n) {
    c[i] = 0.0f; h[i] = 0.0f;
    c16[i] = (_Float16)0.0f; h16[i] = (_Float16)0.0f;
  }
}

// ---------------------------------------------------------------------------
// Merged per-step GEMM kernel. Both recurrent GEMMs depend only on the
// PREVIOUS step's state, so they are independent and run concurrently:
//   blocks [0, GATE_BLOCKS):            g[256,4096] = [h16|x_t] @ BcombT^T (K=1280)
//   blocks [GATE_BLOCKS, +DEC_BLOCKS):  dec[256,1024] = c16 @ WdT^T       (K=1024)
// Each wave computes a 16(M) x 64(N) strip: 4 f32 accumulators reuse one
// A fragment; B fragments are single contiguous 32B v16h loads from the
// pre-transposed [N][K] weights (L2-resident across all 256 steps).
// ---------------------------------------------------------------------------
__global__ void k_step_gemms(const _Float16* __restrict__ h16,
                             const _Float16* __restrict__ x16,
                             const _Float16* __restrict__ BcT,   // [4096][1280]
                             const _Float16* __restrict__ c16,
                             const _Float16* __restrict__ WdT,   // [1024][1024]
                             float* __restrict__ g,
                             float* __restrict__ dec,
                             int t) {
  const int lane = threadIdx.x & 31;
  const int ln   = lane & 15;
  const int grp  = lane >> 4;

  if (blockIdx.x < GATE_BLOCKS) {
    // ---- gate GEMM: M=256, N=4096, K=1280, A = [h | x_t] ----
    const int w  = blockIdx.x * (blockDim.x >> 5) + (threadIdx.x >> 5);
    const int mt = w & 15;
    const int nq = w >> 4;
    const int m0 = mt << 4;
    const int n0 = nq << 6;
    const int mL = m0 + ln;

    const _Float16* __restrict__ hrow = h16 + (size_t)mL * H_;
    const _Float16* __restrict__ xrow = x16 + ((size_t)mL * T_ + t) * DIN;
    const _Float16* __restrict__ b0p  = BcT + (size_t)(n0 + ln) * KC + grp * 16;
    const size_t bs = (size_t)16 * KC;

    v8f acc0 = {}, acc1 = {}, acc2 = {}, acc3 = {};
    for (int k = 0; k < KC; k += 32) {
      // k-tiles of 32 never straddle the H_ boundary (1024 % 32 == 0)
      const _Float16* ap = (k < H_) ? (hrow + k) : (xrow + (k - H_));
      v16h a  = load_a_frag(ap, grp);
      v16h b0 = *(const v16h*)(b0p + k);
      v16h b1 = *(const v16h*)(b0p + bs + k);
      v16h b2 = *(const v16h*)(b0p + 2 * bs + k);
      v16h b3 = *(const v16h*)(b0p + 3 * bs + k);
      acc0 = wmma32(a, b0, acc0);
      acc1 = wmma32(a, b1, acc1);
      acc2 = wmma32(a, b2, acc2);
      acc3 = wmma32(a, b3, acc3);
    }

    alignas(32) float t0[8], t1[8], t2[8], t3[8];
    *(v8f*)t0 = acc0; *(v8f*)t1 = acc1; *(v8f*)t2 = acc2; *(v8f*)t3 = acc3;
    const int row0 = m0 + grp * 8;
#pragma unroll
    for (int r = 0; r < 8; ++r) {
      size_t base = (size_t)(row0 + r) * G4 + n0 + ln;
      g[base]      = t0[r];
      g[base + 16] = t1[r];
      g[base + 32] = t2[r];
      g[base + 48] = t3[r];
    }
  } else {
    // ---- decomp GEMM: M=256, N=1024, K=1024, A = c16 ----
    const int w  = (blockIdx.x - GATE_BLOCKS) * (blockDim.x >> 5) + (threadIdx.x >> 5);
    const int mt = w & 15;
    const int nq = w >> 4;
    const int m0 = mt << 4;
    const int n0 = nq << 6;

    const _Float16* __restrict__ arow = c16 + (size_t)(m0 + ln) * H_;
    const _Float16* __restrict__ b0p  = WdT + (size_t)(n0 + ln) * H_ + grp * 16;
    const size_t bs = (size_t)16 * H_;

    v8f acc0 = {}, acc1 = {}, acc2 = {}, acc3 = {};
    for (int k = 0; k < H_; k += 32) {
      v16h a  = load_a_frag(arow + k, grp);
      v16h b0 = *(const v16h*)(b0p + k);
      v16h b1 = *(const v16h*)(b0p + bs + k);
      v16h b2 = *(const v16h*)(b0p + 2 * bs + k);
      v16h b3 = *(const v16h*)(b0p + 3 * bs + k);
      acc0 = wmma32(a, b0, acc0);
      acc1 = wmma32(a, b1, acc1);
      acc2 = wmma32(a, b2, acc2);
      acc3 = wmma32(a, b3, acc3);
    }

    alignas(32) float t0[8], t1[8], t2[8], t3[8];
    *(v8f*)t0 = acc0; *(v8f*)t1 = acc1; *(v8f*)t2 = acc2; *(v8f*)t3 = acc3;
    const int row0 = m0 + grp * 8;
#pragma unroll
    for (int r = 0; r < 8; ++r) {
      size_t base = (size_t)(row0 + r) * H_ + n0 + ln;
      dec[base]      = t0[r];
      dec[base + 16] = t1[r];
      dec[base + 32] = t2[r];
      dec[base + 48] = t3[r];
    }
  }
}

// ---------------------------------------------------------------------------
// Generic wave-tiled WMMA GEMM (used for the final FC1 layer):
// C[M,N] = A16[M,K] * Bt16[N,K]^T, optional bias + relu epilogue.
// ---------------------------------------------------------------------------
__global__ void k_gemm_nt(const _Float16* __restrict__ A,
                          const _Float16* __restrict__ Bt,
                          float* __restrict__ C,
                          int M, int N, int K,
                          const float* __restrict__ bias, int relu) {
  const int lane = threadIdx.x & 31;
  const int w    = blockIdx.x * (blockDim.x >> 5) + (threadIdx.x >> 5);
  const int mTiles = M >> 4;
  const int mt = w % mTiles;
  const int nq = w / mTiles;
  const int m0 = mt << 4;
  const int n0 = nq << 6;
  if (n0 >= N) return;
  const int ln  = lane & 15;
  const int grp = lane >> 4;

  const _Float16* __restrict__ arow = A + (size_t)(m0 + ln) * K;
  const _Float16* __restrict__ b0p  = Bt + (size_t)(n0 + ln) * K + grp * 16;
  const size_t bs = (size_t)16 * K;

  v8f acc0 = {}, acc1 = {}, acc2 = {}, acc3 = {};
  for (int k = 0; k < K; k += 32) {
    v16h a  = load_a_frag(arow + k, grp);
    v16h b0 = *(const v16h*)(b0p + k);
    v16h b1 = *(const v16h*)(b0p + bs + k);
    v16h b2 = *(const v16h*)(b0p + 2 * bs + k);
    v16h b3 = *(const v16h*)(b0p + 3 * bs + k);
    acc0 = wmma32(a, b0, acc0);
    acc1 = wmma32(a, b1, acc1);
    acc2 = wmma32(a, b2, acc2);
    acc3 = wmma32(a, b3, acc3);
  }

  alignas(32) float t0[8], t1[8], t2[8], t3[8];
  *(v8f*)t0 = acc0; *(v8f*)t1 = acc1; *(v8f*)t2 = acc2; *(v8f*)t3 = acc3;
  const int row0 = m0 + grp * 8;
  float b0v = 0.f, b1v = 0.f, b2v = 0.f, b3v = 0.f;
  if (bias) {
    b0v = bias[n0 + ln];      b1v = bias[n0 + ln + 16];
    b2v = bias[n0 + ln + 32]; b3v = bias[n0 + ln + 48];
  }
#pragma unroll
  for (int r = 0; r < 8; ++r) {
    size_t base = (size_t)(row0 + r) * N + n0 + ln;
    float v0 = t0[r] + b0v, v1 = t1[r] + b1v, v2 = t2[r] + b2v, v3 = t3[r] + b3v;
    if (relu) {
      v0 = fmaxf(v0, 0.f); v1 = fmaxf(v1, 0.f);
      v2 = fmaxf(v2, 0.f); v3 = fmaxf(v3, 0.f);
    }
    C[base] = v0; C[base + 16] = v1; C[base + 32] = v2; C[base + 48] = v3;
  }
}

// ---------------------------------------------------------------------------
// Per-step pointwise epilogue: time-decay decomposition + LSTM gates.
// Produces f32 state (c,h) and f16 copies for next step's WMMA operands.
// ---------------------------------------------------------------------------
__global__ void k_step_pointwise(const float* __restrict__ dec,
                                 const float* __restrict__ g,
                                 const float* __restrict__ elapsed, int t,
                                 const float* __restrict__ bd,
                                 const float* __restrict__ b_decomp,
                                 const float* __restrict__ b_lin,
                                 const float* __restrict__ b_extra,
                                 float* __restrict__ c, float* __restrict__ h,
                                 _Float16* __restrict__ c16, _Float16* __restrict__ h16) {
  int idx = blockIdx.x * blockDim.x + threadIdx.x;
  if (idx >= B_ * H_) return;
  int b = idx >> 10;        // H_ == 1024
  int j = idx & (H_ - 1);

  float dt  = elapsed[(size_t)b * T_ + t];
  float Td  = 1.0f / logf(dt + 2.7183f);
  float cst = tanhf(dec[idx] + bd[j] + b_decomp[j]);
  float cpre = c[idx] - cst + Td * cst;

  const float* __restrict__ gb = g + (size_t)b * G4;
  float gi = gb[j]            + b_lin[j]            + b_extra[j];
  float gf = gb[H_ + j]       + b_lin[H_ + j]       + b_extra[H_ + j];
  float go = gb[2 * H_ + j]   + b_lin[2 * H_ + j]   + b_extra[2 * H_ + j];
  float gc = gb[3 * H_ + j]   + b_lin[3 * H_ + j]   + b_extra[3 * H_ + j];

  float cn = sigm(gf) * cpre + sigm(gi) * tanhf(gc);
  float hn = sigm(go) * tanhf(cn);

  c[idx] = cn;  h[idx] = hn;
  c16[idx] = (_Float16)cn;
  h16[idx] = (_Float16)hn;
}

// ---------------------------------------------------------------------------
// Tiny final layer: out[256,24] = fc1out[256,512] @ fco_w[512,24] + fco_b
// ---------------------------------------------------------------------------
__global__ void k_fco(const float* __restrict__ f1, const float* __restrict__ W,
                      const float* __restrict__ bias, float* __restrict__ out) {
  int idx = blockIdx.x * blockDim.x + threadIdx.x;
  if (idx >= B_ * DOUT) return;
  int b = idx / DOUT;
  int d = idx % DOUT;
  const float* __restrict__ row = f1 + (size_t)b * FC_;
  float acc = bias[d];
  for (int k = 0; k < FC_; ++k)
    acc = fmaf(row[k], W[(size_t)k * DOUT + d], acc);
  out[idx] = acc;
}

// ---------------------------------------------------------------------------
extern "C" void kernel_launch(void* const* d_in, const int* in_sizes, int n_in,
                              void* d_out, int out_size, void* d_ws, size_t ws_size,
                              hipStream_t stream) {
  (void)in_sizes; (void)n_in; (void)out_size; (void)ws_size;

  const float* x        = (const float*)d_in[0];   // [B,T,DIN]
  const float* et       = (const float*)d_in[1];   // [B,T]
  const float* Wx       = (const float*)d_in[2];   // [DIN,4H]
  const float* Ux       = (const float*)d_in[3];   // [H,4H]
  const float* b_lin    = (const float*)d_in[4];   // [4H]
  const float* b_extra  = (const float*)d_in[5];   // [4H]
  const float* Wd       = (const float*)d_in[6];   // [H,H]
  const float* bd       = (const float*)d_in[7];   // [H]
  const float* b_decomp = (const float*)d_in[8];   // [H]
  const float* fc1_w    = (const float*)d_in[9];   // [H,FC]
  const float* fc1_b    = (const float*)d_in[10];  // [FC]
  const float* fco_w    = (const float*)d_in[11];  // [FC,DOUT]
  const float* fco_b    = (const float*)d_in[12];  // [DOUT]
  float* out = (float*)d_out;                      // [B,DOUT]

  char* p = (char*)d_ws;
  auto alloc = [&](size_t bytes) -> char* {
    char* r = p;
    p += (bytes + 255) & ~(size_t)255;
    return r;
  };
  _Float16* x16 = (_Float16*)alloc((size_t)B_ * T_ * DIN * 2);  // 32 MB
  _Float16* BcT = (_Float16*)alloc((size_t)G4 * KC * 2);        // 10 MB
  _Float16* WdT = (_Float16*)alloc((size_t)H_ * H_ * 2);        //  2 MB
  _Float16* f1T = (_Float16*)alloc((size_t)FC_ * H_ * 2);       //  1 MB
  float*    c   = (float*)alloc((size_t)B_ * H_ * 4);
  float*    h   = (float*)alloc((size_t)B_ * H_ * 4);
  _Float16* c16 = (_Float16*)alloc((size_t)B_ * H_ * 2);
  _Float16* h16 = (_Float16*)alloc((size_t)B_ * H_ * 2);
  float*    dec = (float*)alloc((size_t)B_ * H_ * 4);
  float*    g   = (float*)alloc((size_t)B_ * G4 * 4);
  float*    f1o = (float*)alloc((size_t)B_ * FC_ * 4);

  // One-time conversions / weight layout transforms
  {
    int n = B_ * T_ * DIN;
    k_f32_to_f16<<<(n + 255) / 256, 256, 0, stream>>>(x, x16, n);
  }
  k_build_bcomb<<<(G4 * KC + 255) / 256, 256, 0, stream>>>(Ux, Wx, BcT);
  k_transpose_f16<<<(H_ * H_ + 255) / 256, 256, 0, stream>>>(Wd, WdT, H_, H_);
  k_transpose_f16<<<(H_ * FC_ + 255) / 256, 256, 0, stream>>>(fc1_w, f1T, H_, FC_);
  k_zero_state<<<(B_ * H_ + 255) / 256, 256, 0, stream>>>(c, h, c16, h16, B_ * H_);

  // Recurrence: 256 sequential steps, 2 dependent launches per step.
  for (int t = 0; t < T_; ++t) {
    // Both recurrent GEMMs in one grid (they only read previous-step state):
    //   128 gate blocks + 32 decomp blocks = 160 blocks, 1280 waves.
    k_step_gemms<<<GATE_BLOCKS + DEC_BLOCKS, 256, 0, stream>>>(
        h16, x16, BcT, c16, WdT, g, dec, t);
    k_step_pointwise<<<(B_ * H_ + 255) / 256, 256, 0, stream>>>(
        dec, g, et, t, bd, b_decomp, b_lin, b_extra, c, h, c16, h16);
  }

  // fc1: relu(h @ fc1_w + fc1_b)  (M=256,N=512,K=1024) -> 128 waves -> 16 blocks
  k_gemm_nt<<<16, 256, 0, stream>>>(h16, f1T, f1o, B_, FC_, H_, fc1_b, 1);
  // final projection (tiny)
  k_fco<<<(B_ * DOUT + 255) / 256, 256, 0, stream>>>(f1o, fco_w, fco_b, out);
}
